// FreeConvNetwork_29231547416691
// MI455X (gfx1250) — matrix operand order repaired and checked by reading, not compiled, not run on hardware
//
#include <hip/hip_runtime.h>
#include <hip/hip_bf16.h>
#include <math.h>

typedef __attribute__((ext_vector_type(16))) _Float16 v16h;
typedef __attribute__((ext_vector_type(8)))  _Float16 v8h;
typedef __attribute__((ext_vector_type(8)))  float    v8f;

// ---------------------------------------------------------------------------
// Pack an fp32 weight tensor W[Cout][Kd][P] (locally-connected: per-position
// weights; FC: P==1) into CDNA5 WMMA B-fragment layout, f16, K zero-padded to
// ksteps*32.  Bp layout: [p][nt][ks][lane(32)][j(16)] halves.
// B-fragment (32x16 f16), contiguous k order used here (A matches): lane L holds column N = (L&15),
//   half j -> K = ks*32 + (L>>4)*16 + j.
// ---------------------------------------------------------------------------
__global__ void pack_b_kernel(const float* __restrict__ W, _Float16* __restrict__ Bp,
                              int Kd, int P, int ntiles, int ksteps, int total)
{
    int idx = blockIdx.x * blockDim.x + threadIdx.x;
    if (idx >= total) return;
    int j    = idx & 15;
    int lane = (idx >> 4) & 31;
    int rest = idx >> 9;
    int ks = rest % ksteps; rest /= ksteps;
    int nt = rest % ntiles;
    int p  = rest / ntiles;
    int nr = lane & 15, hi = lane >> 4;
    int o = nt * 16 + nr;
    int k = ks * 32 + hi * 16 + j;
    float v = 0.0f;
    if (k < Kd) v = W[((size_t)o * Kd + k) * P + p];
    Bp[idx] = (_Float16)v;
}

// ---------------------------------------------------------------------------
// im2col for layer1: x fp32 (N,1,27,27), stride 2, 3x3 -> A[p][n][32] f16
// (k = ky*3+kx for k<9, zero pad to 32).  p = oy*13+ox.
// ---------------------------------------------------------------------------
__global__ void im2col1_kernel(const float* __restrict__ x, _Float16* __restrict__ A, int Nrows)
{
    int n = blockIdx.x * blockDim.x + threadIdx.x;
    int p = blockIdx.y;
    if (n >= Nrows) return;
    int oy = p / 13, ox = p % 13;
    const float* xp = x + (size_t)n * 729 + (oy * 2) * 27 + (ox * 2);
    _Float16* ap = A + ((size_t)p * Nrows + n) * 32;
    #pragma unroll
    for (int ky = 0; ky < 3; ++ky)
        #pragma unroll
        for (int kx = 0; kx < 3; ++kx)
            ap[ky * 3 + kx] = (_Float16)xp[ky * 27 + kx];
    #pragma unroll
    for (int k = 9; k < 32; ++k) ap[k] = (_Float16)0.0f;
}

// im2col for layer2: h1 f16 (N,16,13,13), stride 2, 3x3 -> A[p][n][160]
// (k = c*9+ky*3+kx for k<144, zero pad).  p = oy*6+ox.
__global__ void im2col2_kernel(const _Float16* __restrict__ h1, _Float16* __restrict__ A, int Nrows)
{
    int n = blockIdx.x * blockDim.x + threadIdx.x;
    int p = blockIdx.y;
    if (n >= Nrows) return;
    int oy = p / 6, ox = p % 6;
    const _Float16* hp = h1 + (size_t)n * 2704 + (oy * 2) * 13 + (ox * 2);
    _Float16* ap = A + ((size_t)p * Nrows + n) * 160;
    int k = 0;
    for (int c = 0; c < 16; ++c)
        #pragma unroll
        for (int ky = 0; ky < 3; ++ky)
            #pragma unroll
            for (int kx = 0; kx < 3; ++kx)
                ap[k++] = hp[c * 169 + ky * 13 + kx];
    for (; k < 160; ++k) ap[k] = (_Float16)0.0f;
}

// im2col for layer3: h2 f16 (N,32,6,6), stride 1, 3x3 -> A[p][n][288] (exact).
// p = oy*4+ox.
__global__ void im2col3_kernel(const _Float16* __restrict__ h2, _Float16* __restrict__ A, int Nrows)
{
    int n = blockIdx.x * blockDim.x + threadIdx.x;
    int p = blockIdx.y;
    if (n >= Nrows) return;
    int oy = p / 4, ox = p % 4;
    const _Float16* hp = h2 + (size_t)n * 1152 + oy * 6 + ox;
    _Float16* ap = A + ((size_t)p * Nrows + n) * 288;
    int k = 0;
    for (int c = 0; c < 32; ++c)
        #pragma unroll
        for (int ky = 0; ky < 3; ++ky)
            #pragma unroll
            for (int kx = 0; kx < 3; ++kx)
                ap[k++] = hp[c * 36 + ky * 6 + kx];
}

// ---------------------------------------------------------------------------
// Generic batched WMMA GEMM with register blocking:
//   D[p] = relu( A[p] (Nrows x Kp, f16) * B[p] (pre-packed fragments) + bias )
// Each wave computes a 16(M) x 16*NT(N) tile: one A fragment is reused across
// NT B fragments -> NT independent v_wmma chains per k-step (hides XDL and
// load latency; 4x less A traffic at NT=4).
// Block = 128 threads (4 waves stacked along M, M-block = 64).
// grid = (Nrows/64, ntiles_total/NT, P).
// Output element (m, o) stored at out[m*sN + o*sO + p*sP].
// ---------------------------------------------------------------------------
template<int NT>
__global__ void wmma_gemm_kernel(const _Float16* __restrict__ A,
                                 const _Float16* __restrict__ Bp,
                                 const float*    __restrict__ bias,
                                 _Float16*       __restrict__ out,
                                 int Nrows, int Kp, int ksteps, int ntiles_total,
                                 int biasStrideO, long sN, long sO, long sP)
{
    const int lane = threadIdx.x & 31;
    const int wave = threadIdx.x >> 5;
    const int m0  = blockIdx.x * 64 + wave * 16;
    const int nt0 = blockIdx.y * NT;
    const int p   = blockIdx.z;
    const int nr = lane & 15, hi = lane >> 4;

    const _Float16* Ap = A + (size_t)p * Nrows * Kp + (size_t)(m0 + nr) * Kp + hi * 16;
    const _Float16* Bt[NT];
    v8f acc[NT];
    #pragma unroll
    for (int t = 0; t < NT; ++t) {
        Bt[t] = Bp + (((size_t)p * ntiles_total + nt0 + t) * (size_t)ksteps) * 512 + lane * 16;
        const float bv = bias[((nt0 + t) * 16 + nr) * biasStrideO + p];
        #pragma unroll
        for (int r = 0; r < 8; ++r) acc[t][r] = bv;
    }

    for (int ks = 0; ks < ksteps; ++ks) {
        const _Float16* pa = Ap + ks * 32;
        if (ks + 1 < ksteps) __builtin_prefetch(pa + 32, 0, 3);
        v16h a = *(const v16h*)(pa);        // K = ks*32 + hi*16 + [0..15]
        v16h b[NT];
        #pragma unroll
        for (int t = 0; t < NT; ++t) b[t] = *(const v16h*)(Bt[t] + ks * 512);
        #pragma unroll
        for (int t = 0; t < NT; ++t)
            acc[t] = __builtin_amdgcn_wmma_f32_16x16x32_f16(
                /*neg_a=*/false, a, /*neg_b=*/false, b[t],
                /*c_mod=*/(short)0, acc[t], /*reuse_a=*/false, /*reuse_b=*/false);
    }

    #pragma unroll
    for (int t = 0; t < NT; ++t) {
        const int o = (nt0 + t) * 16 + nr;
        #pragma unroll
        for (int r = 0; r < 8; ++r) {
            int m = m0 + r + hi * 8;         // C/D layout: VGPR r, hi-half lanes -> M = r+8
            float v = acc[t][r];
            v = v > 0.0f ? v : 0.0f;         // fused ReLU (all four GEMM layers use it)
            out[(size_t)m * sN + (size_t)o * sO + (size_t)p * sP] = (_Float16)v;
        }
    }
}

// ---------------------------------------------------------------------------
// FC2 (512 -> 10) + softmax, one wave32 per sample, fp32 accumulate.
// ---------------------------------------------------------------------------
__global__ void fc2_softmax_kernel(const _Float16* __restrict__ h4,
                                   const float* __restrict__ W2,
                                   const float* __restrict__ B2,
                                   float* __restrict__ out, int Nrows)
{
    const int lane = threadIdx.x & 31;
    const int wave = threadIdx.x >> 5;
    const int n = blockIdx.x * (blockDim.x >> 5) + wave;
    if (n >= Nrows) return;

    float acc[10];
    #pragma unroll
    for (int j = 0; j < 10; ++j) acc[j] = 0.0f;

    for (int k = lane; k < 512; k += 32) {
        float hv = (float)h4[(size_t)n * 512 + k];
        #pragma unroll
        for (int j = 0; j < 10; ++j) acc[j] += hv * W2[j * 512 + k];
    }
    #pragma unroll
    for (int j = 0; j < 10; ++j)
        #pragma unroll
        for (int off = 16; off > 0; off >>= 1)
            acc[j] += __shfl_xor(acc[j], off, 32);

    if (lane == 0) {
        float mx = -1e30f;
        #pragma unroll
        for (int j = 0; j < 10; ++j) { acc[j] += B2[j]; mx = fmaxf(mx, acc[j]); }
        float s = 0.0f, e[10];
        #pragma unroll
        for (int j = 0; j < 10; ++j) { e[j] = expf(acc[j] - mx); s += e[j]; }
        float inv = 1.0f / s;
        #pragma unroll
        for (int j = 0; j < 10; ++j) out[(size_t)n * 10 + j] = e[j] * inv;
    }
}

// ---------------------------------------------------------------------------
extern "C" void kernel_launch(void* const* d_in, const int* in_sizes, int n_in,
                              void* d_out, int out_size, void* d_ws, size_t ws_size,
                              hipStream_t stream) {
    const float* x  = (const float*)d_in[0];   // (N,1,27,27)
    const float* w1 = (const float*)d_in[1];   // (16,1,9,13,13)
    const float* b1 = (const float*)d_in[2];   // (16,13,13)
    const float* w2 = (const float*)d_in[3];   // (32,16,9,6,6)
    const float* b2 = (const float*)d_in[4];   // (32,6,6)
    const float* w3 = (const float*)d_in[5];   // (64,32,9,4,4)
    const float* b3 = (const float*)d_in[6];   // (64,4,4)
    const float* W1 = (const float*)d_in[7];   // (512,1024)
    const float* B1 = (const float*)d_in[8];   // (512)
    const float* W2 = (const float*)d_in[9];   // (10,512)
    const float* B2 = (const float*)d_in[10];  // (10)
    float* out = (float*)d_out;

    const int N = in_sizes[0] / 729;           // 8192 (assume divisible by 64)

    // --- workspace arena (256B aligned slots, reused across layers) ---
    char* base = (char*)d_ws;
    size_t off = 0;
    auto arena = [&](size_t bytes) -> char* {
        char* p = base + off;
        off = (off + bytes + 255) & ~(size_t)255;
        return p;
    };
    _Float16* Bp1 = (_Float16*)arena((size_t)169 * 1 * 1  * 512 * 2);
    _Float16* Bp2 = (_Float16*)arena((size_t)36  * 2 * 5  * 512 * 2);
    _Float16* Bp3 = (_Float16*)arena((size_t)16  * 4 * 9  * 512 * 2);
    _Float16* BpF = (_Float16*)arena((size_t)1   * 32 * 32 * 512 * 2);
    // slotA holds A1 (N*169*32), then A2 (N*36*160), then A3 (N*16*288) halves
    _Float16* slotA = (_Float16*)arena((size_t)N * 5760 * 2);
    // slotH holds h1 (N*2704), then h2 (N*1152), then h3 (N*1024) halves
    _Float16* slotH = (_Float16*)arena((size_t)N * 2704 * 2);
    _Float16* h4    = (_Float16*)arena((size_t)N * 512 * 2);
    (void)ws_size; (void)n_in; (void)out_size;

    // --- pack all weight matrices into WMMA B-fragment layout (f16) ---
    {
        int t1 = 169 * 1 * 1 * 512;
        pack_b_kernel<<<(t1 + 255) / 256, 256, 0, stream>>>(w1, Bp1, 9,    169, 1,  1,  t1);
        int t2 = 36 * 2 * 5 * 512;
        pack_b_kernel<<<(t2 + 255) / 256, 256, 0, stream>>>(w2, Bp2, 144,  36,  2,  5,  t2);
        int t3 = 16 * 4 * 9 * 512;
        pack_b_kernel<<<(t3 + 255) / 256, 256, 0, stream>>>(w3, Bp3, 288,  16,  4,  9,  t3);
        int tF = 32 * 32 * 512;
        pack_b_kernel<<<(tF + 255) / 256, 256, 0, stream>>>(W1, BpF, 1024, 1,   32, 32, tF);
    }

    dim3 blk128(128), blk256(256);

    // --- Layer 1: im2col + WMMA GEMM (Cout=16 -> NT=1) -> h1 (N,16,13,13) f16
    im2col1_kernel<<<dim3((N + 255) / 256, 169), blk256, 0, stream>>>(x, slotA, N);
    wmma_gemm_kernel<1><<<dim3(N / 64, 1, 169), blk128, 0, stream>>>(
        slotA, Bp1, b1, slotH, N, 32, 1, 1, /*biasStrideO=*/169,
        /*sN=*/2704, /*sO=*/169, /*sP=*/1);

    // --- Layer 2: im2col + WMMA GEMM (Cout=32 -> NT=2) -> h2 (N,32,6,6) f16
    im2col2_kernel<<<dim3((N + 255) / 256, 36), blk256, 0, stream>>>(slotH, slotA, N);
    wmma_gemm_kernel<2><<<dim3(N / 64, 1, 36), blk128, 0, stream>>>(
        slotA, Bp2, b2, slotH, N, 160, 5, 2, /*biasStrideO=*/36,
        /*sN=*/1152, /*sO=*/36, /*sP=*/1);

    // --- Layer 3: im2col + WMMA GEMM (Cout=64 -> NT=4) -> h3 flat (N,1024) f16
    im2col3_kernel<<<dim3((N + 255) / 256, 16), blk256, 0, stream>>>(slotH, slotA, N);
    wmma_gemm_kernel<4><<<dim3(N / 64, 1, 16), blk128, 0, stream>>>(
        slotA, Bp3, b3, slotH, N, 288, 9, 4, /*biasStrideO=*/16,
        /*sN=*/1024, /*sO=*/16, /*sP=*/1);

    // --- FC1: h3 (N x 1024) * W1^T -> h4 (N x 512), ReLU  (NT=4, grid.y=8)
    wmma_gemm_kernel<4><<<dim3(N / 64, 8, 1), blk128, 0, stream>>>(
        slotH, BpF, B1, h4, N, 1024, 32, 32, /*biasStrideO=*/1,
        /*sN=*/512, /*sO=*/1, /*sP=*/0);

    // --- FC2 + softmax -> out (N x 10) fp32 ---
    fc2_softmax_kernel<<<dim3(N / 8), blk256, 0, stream>>>(h4, W2, B2, out, N);
}
